// TransformerBlock_73615739453757
// MI455X (gfx1250) — matrix operand order repaired and checked by reading, not compiled, not run on hardware
//
#include <hip/hip_runtime.h>
#include <cstdint>
#include <cstddef>

// ---------------------------------------------------------------------------
// CDNA5 (gfx1250) types & helpers
// ---------------------------------------------------------------------------
typedef __bf16 bf16_t;
typedef __attribute__((ext_vector_type(16))) __bf16 v16bf;
typedef __attribute__((ext_vector_type(8)))  __bf16 v8bf;
typedef __attribute__((ext_vector_type(8)))  float  v8f;

#define WMMA_BF16(a, b, c) \
  __builtin_amdgcn_wmma_f32_16x16x32_bf16(false, (a), false, (b), (short)0, (c), false, false)

__device__ __forceinline__ bf16_t f2bf(float f) {
  union { float f; uint32_t u; } in; in.f = f;
  uint32_t u = in.u;
  u += 0x7FFFu + ((u >> 16) & 1u);          // round-to-nearest-even
  union { uint16_t s; bf16_t h; } out; out.s = (uint16_t)(u >> 16);
  return out.h;
}

__device__ __forceinline__ v16bf cat16(v8bf lo, v8bf hi) {
  return __builtin_shufflevector(lo, hi, 0,1,2,3,4,5,6,7,8,9,10,11,12,13,14,15);
}

__device__ __forceinline__ float siluf(float x) { return x / (1.f + __expf(-x)); }

#define BK  32
#define BKP 48   // padded LDS K-stride (elements); 96B rows keep 16B alignment

// ---------------------------------------------------------------------------
// Generic GEMM: C[M,N] = A[M,K] (fp32) x B[K,N] (fp32), bf16 WMMA, fp32 accum
// Block tile 128x128x32, 256 threads = 8 waves (2x4), wave tile 64x32.
// ---------------------------------------------------------------------------
__global__ __launch_bounds__(256) void gemm_nn(
    const float* __restrict__ A, const float* __restrict__ B, float* __restrict__ C,
    int M, int N, int K, int lda, int ldb, int ldc)
{
  __shared__ __attribute__((aligned(16))) bf16_t As[128 * BKP];
  __shared__ __attribute__((aligned(16))) bf16_t Bs[128 * BKP];

  const int tid  = threadIdx.x;
  const int m0   = blockIdx.y * 128;
  const int n0   = blockIdx.x * 128;
  const int wave = tid >> 5, lane = tid & 31;
  const int lm   = lane & 15, lhk = lane >> 4;
  const int wm0  = (wave >> 2) * 64;
  const int wn0  = (wave & 3) * 32;

  v8f acc[4][2];
#pragma unroll
  for (int i = 0; i < 4; ++i)
#pragma unroll
    for (int j = 0; j < 2; ++j)
#pragma unroll
      for (int q = 0; q < 8; ++q) acc[i][j][q] = 0.f;

  for (int k0 = 0; k0 < K; k0 += BK) {
    // ---- stage A tile (128x32 fp32 -> bf16) ----
#pragma unroll
    for (int it = 0; it < 4; ++it) {
      int f4 = tid + it * 256;           // 0..1023
      int r  = f4 >> 3;
      int c  = (f4 & 7) * 4;
      float4 v; v.x = v.y = v.z = v.w = 0.f;
      if (m0 + r < M) v = *(const float4*)(A + (size_t)(m0 + r) * lda + k0 + c);
      bf16_t* d = &As[r * BKP + c];
      d[0] = f2bf(v.x); d[1] = f2bf(v.y); d[2] = f2bf(v.z); d[3] = f2bf(v.w);
    }
    // ---- stage B tile (32x128 fp32 -> bf16, transposed [n][k]) ----
#pragma unroll
    for (int it = 0; it < 4; ++it) {
      int f4 = tid + it * 256;
      int kr = f4 >> 5;
      int c  = (f4 & 31) * 4;
      float4 v; v.x = v.y = v.z = v.w = 0.f;
      if (n0 + c < N) v = *(const float4*)(B + (size_t)(k0 + kr) * ldb + n0 + c);
      Bs[(c + 0) * BKP + kr] = f2bf(v.x);
      Bs[(c + 1) * BKP + kr] = f2bf(v.y);
      Bs[(c + 2) * BKP + kr] = f2bf(v.z);
      Bs[(c + 3) * BKP + kr] = f2bf(v.w);
    }
    // prefetch next A tile rows into L2 (global_prefetch_b8)
    if (k0 + BK < K) {
      int pr = tid >> 1;
      if (m0 + pr < M) __builtin_prefetch(A + (size_t)(m0 + pr) * lda + k0 + BK, 0, 0);
    }
    __syncthreads();

    v16bf afrag[4], bfrag[2];
#pragma unroll
    for (int i = 0; i < 4; ++i) {
      const bf16_t* p = &As[(wm0 + i * 16 + lm) * BKP + lhk * 8];
      afrag[i] = cat16(*(const v8bf*)p, *(const v8bf*)(p + 16));
    }
#pragma unroll
    for (int j = 0; j < 2; ++j) {
      const bf16_t* p = &Bs[(wn0 + j * 16 + lm) * BKP + lhk * 8];
      bfrag[j] = cat16(*(const v8bf*)p, *(const v8bf*)(p + 16));
    }
#pragma unroll
    for (int i = 0; i < 4; ++i)
#pragma unroll
      for (int j = 0; j < 2; ++j)
        acc[i][j] = WMMA_BF16(afrag[i], bfrag[j], acc[i][j]);
    __syncthreads();
  }

#pragma unroll
  for (int i = 0; i < 4; ++i)
#pragma unroll
    for (int j = 0; j < 2; ++j)
#pragma unroll
      for (int q = 0; q < 8; ++q) {
        int row = m0 + wm0 + i * 16 + lhk * 8 + q;
        int col = n0 + wn0 + j * 16 + lm;
        if (row < M && col < N) C[(size_t)row * ldc + col] = acc[i][j][q];
      }
}

// ---------------------------------------------------------------------------
// MoE dual GEMM: H1[row] = silu(X[g]·W1e) * (X[g]·W3e), gathered rows, bf16 out
// Block tile 64x128x32, wave tile 32x32.
// ---------------------------------------------------------------------------
__global__ __launch_bounds__(256) void moe_gemm1(
    const float* __restrict__ X, const float* __restrict__ W1, const float* __restrict__ W3,
    bf16_t* __restrict__ H1,
    const int* __restrict__ rows, const int* __restrict__ counts, const int* __restrict__ offsets,
    int M, int K, int N, size_t strideB)
{
  const int e    = blockIdx.z;
  const int cnt  = counts ? counts[e] : M;
  const int base = offsets ? offsets[e] : 0;
  const int m0   = blockIdx.y * 64;
  if (m0 >= cnt) return;
  const int n0 = blockIdx.x * 128;
  const float* B1 = W1 + strideB * e;
  const float* B3 = W3 + strideB * e;

  __shared__ __attribute__((aligned(16))) bf16_t As [64  * BKP];
  __shared__ __attribute__((aligned(16))) bf16_t B1s[128 * BKP];
  __shared__ __attribute__((aligned(16))) bf16_t B3s[128 * BKP];

  const int tid  = threadIdx.x;
  const int wave = tid >> 5, lane = tid & 31;
  const int lm   = lane & 15, lhk = lane >> 4;
  const int wm0  = (wave >> 2) * 32;
  const int wn0  = (wave & 3) * 32;

  v8f acc1[2][2], acc3[2][2];
#pragma unroll
  for (int i = 0; i < 2; ++i)
#pragma unroll
    for (int j = 0; j < 2; ++j)
#pragma unroll
      for (int q = 0; q < 8; ++q) { acc1[i][j][q] = 0.f; acc3[i][j][q] = 0.f; }

  for (int k0 = 0; k0 < K; k0 += BK) {
#pragma unroll
    for (int it = 0; it < 2; ++it) {           // A tile 64x32
      int f4 = tid + it * 256;                 // 0..511
      int r  = f4 >> 3;
      int c  = (f4 & 7) * 4;
      float4 v; v.x = v.y = v.z = v.w = 0.f;
      if (m0 + r < cnt) {
        int ar = rows ? rows[base + m0 + r] : (m0 + r);
        v = *(const float4*)(X + (size_t)ar * K + k0 + c);
      }
      bf16_t* d = &As[r * BKP + c];
      d[0] = f2bf(v.x); d[1] = f2bf(v.y); d[2] = f2bf(v.z); d[3] = f2bf(v.w);
    }
#pragma unroll
    for (int it = 0; it < 4; ++it) {           // B1/B3 tiles 32x128, transposed
      int f4 = tid + it * 256;
      int kr = f4 >> 5;
      int c  = (f4 & 31) * 4;
      float4 a = *(const float4*)(B1 + (size_t)(k0 + kr) * N + n0 + c);
      float4 b = *(const float4*)(B3 + (size_t)(k0 + kr) * N + n0 + c);
      B1s[(c + 0) * BKP + kr] = f2bf(a.x); B1s[(c + 1) * BKP + kr] = f2bf(a.y);
      B1s[(c + 2) * BKP + kr] = f2bf(a.z); B1s[(c + 3) * BKP + kr] = f2bf(a.w);
      B3s[(c + 0) * BKP + kr] = f2bf(b.x); B3s[(c + 1) * BKP + kr] = f2bf(b.y);
      B3s[(c + 2) * BKP + kr] = f2bf(b.z); B3s[(c + 3) * BKP + kr] = f2bf(b.w);
    }
    __syncthreads();

    v16bf af[2], b1f[2], b3f[2];
#pragma unroll
    for (int i = 0; i < 2; ++i) {
      const bf16_t* p = &As[(wm0 + i * 16 + lm) * BKP + lhk * 8];
      af[i] = cat16(*(const v8bf*)p, *(const v8bf*)(p + 16));
    }
#pragma unroll
    for (int j = 0; j < 2; ++j) {
      const bf16_t* p1 = &B1s[(wn0 + j * 16 + lm) * BKP + lhk * 8];
      const bf16_t* p3 = &B3s[(wn0 + j * 16 + lm) * BKP + lhk * 8];
      b1f[j] = cat16(*(const v8bf*)p1, *(const v8bf*)(p1 + 16));
      b3f[j] = cat16(*(const v8bf*)p3, *(const v8bf*)(p3 + 16));
    }
#pragma unroll
    for (int i = 0; i < 2; ++i)
#pragma unroll
      for (int j = 0; j < 2; ++j) {
        acc1[i][j] = WMMA_BF16(af[i], b1f[j], acc1[i][j]);
        acc3[i][j] = WMMA_BF16(af[i], b3f[j], acc3[i][j]);
      }
    __syncthreads();
  }

#pragma unroll
  for (int i = 0; i < 2; ++i)
#pragma unroll
    for (int j = 0; j < 2; ++j)
#pragma unroll
      for (int q = 0; q < 8; ++q) {
        int mrow = m0 + wm0 + i * 16 + lhk * 8 + q;
        if (mrow < cnt) {
          int col = n0 + wn0 + j * 16 + lm;
          float h = siluf(acc1[i][j][q]) * acc3[i][j][q];
          H1[(size_t)(base + mrow) * N + col] = f2bf(h);
        }
      }
}

// ---------------------------------------------------------------------------
// MoE down GEMM with scatter-accumulate: out[rowOut[m]] += scale[m]*(H1·W2e)
// A is bf16 (no conversion), block tile 128x128x32.
// ---------------------------------------------------------------------------
__global__ __launch_bounds__(256) void moe_gemm2(
    const bf16_t* __restrict__ H1, const float* __restrict__ W2, float* __restrict__ out,
    const int* __restrict__ rows_out, const float* __restrict__ scales,
    const int* __restrict__ counts, const int* __restrict__ offsets,
    int M, int K, int N, size_t strideB)
{
  const int e    = blockIdx.z;
  const int cnt  = counts ? counts[e] : M;
  const int base = offsets ? offsets[e] : 0;
  const int m0   = blockIdx.y * 128;
  if (m0 >= cnt) return;
  const int n0 = blockIdx.x * 128;
  const float* B = W2 + strideB * e;

  __shared__ __attribute__((aligned(16))) bf16_t As[128 * BKP];
  __shared__ __attribute__((aligned(16))) bf16_t Bs[128 * BKP];

  const int tid  = threadIdx.x;
  const int wave = tid >> 5, lane = tid & 31;
  const int lm   = lane & 15, lhk = lane >> 4;
  const int wm0  = (wave >> 2) * 64;
  const int wn0  = (wave & 3) * 32;

  v8f acc[4][2];
#pragma unroll
  for (int i = 0; i < 4; ++i)
#pragma unroll
    for (int j = 0; j < 2; ++j)
#pragma unroll
      for (int q = 0; q < 8; ++q) acc[i][j][q] = 0.f;

  for (int k0 = 0; k0 < K; k0 += BK) {
#pragma unroll
    for (int it = 0; it < 2; ++it) {           // A tile 128x32 bf16: 16B chunks
      int cid = tid + it * 256;                // 0..511
      int r   = cid >> 2;
      int c   = (cid & 3) * 8;
      v8bf v;
#pragma unroll
      for (int q = 0; q < 8; ++q) v[q] = f2bf(0.f);
      if (m0 + r < cnt) v = *(const v8bf*)(H1 + (size_t)(base + m0 + r) * K + k0 + c);
      *(v8bf*)&As[r * BKP + c] = v;
    }
#pragma unroll
    for (int it = 0; it < 4; ++it) {           // B tile 32x128, transposed
      int f4 = tid + it * 256;
      int kr = f4 >> 5;
      int c  = (f4 & 31) * 4;
      float4 v = *(const float4*)(B + (size_t)(k0 + kr) * N + n0 + c);
      Bs[(c + 0) * BKP + kr] = f2bf(v.x); Bs[(c + 1) * BKP + kr] = f2bf(v.y);
      Bs[(c + 2) * BKP + kr] = f2bf(v.z); Bs[(c + 3) * BKP + kr] = f2bf(v.w);
    }
    __syncthreads();

    v16bf afrag[4], bfrag[2];
#pragma unroll
    for (int i = 0; i < 4; ++i) {
      const bf16_t* p = &As[(wm0 + i * 16 + lm) * BKP + lhk * 8];
      afrag[i] = cat16(*(const v8bf*)p, *(const v8bf*)(p + 16));
    }
#pragma unroll
    for (int j = 0; j < 2; ++j) {
      const bf16_t* p = &Bs[(wn0 + j * 16 + lm) * BKP + lhk * 8];
      bfrag[j] = cat16(*(const v8bf*)p, *(const v8bf*)(p + 16));
    }
#pragma unroll
    for (int i = 0; i < 4; ++i)
#pragma unroll
      for (int j = 0; j < 2; ++j)
        acc[i][j] = WMMA_BF16(afrag[i], bfrag[j], acc[i][j]);
    __syncthreads();
  }

#pragma unroll
  for (int i = 0; i < 4; ++i)
#pragma unroll
    for (int j = 0; j < 2; ++j)
#pragma unroll
      for (int q = 0; q < 8; ++q) {
        int mrow = m0 + wm0 + i * 16 + lhk * 8 + q;
        if (mrow < cnt) {
          int col = n0 + wn0 + j * 16 + lm;
          int ro  = rows_out ? rows_out[base + mrow] : mrow;
          float sc = scales ? scales[base + mrow] : 1.f;
          atomicAdd(&out[(size_t)ro * N + col], acc[i][j][q] * sc);
        }
      }
}

// ---------------------------------------------------------------------------
// Flash-style MLA attention. Block = (q-tile 64, head, batch), 128 thr = 4 waves.
// Per wave: 16 query rows; streams 32-key tiles, online softmax, WMMA everywhere.
// q: [T,16,192] (roped, pre-scaled here), kvp: [T,16,256] (nope|v), kpe: [T,64]
// out: [T, 16*128]
// ---------------------------------------------------------------------------
__global__ __launch_bounds__(128) void mla_attention(
    const float* __restrict__ q, const float* __restrict__ kvp,
    const float* __restrict__ kpe, float* __restrict__ out, int S)
{
  __shared__ __attribute__((aligned(16))) bf16_t Qs[64 * 192];
  __shared__ __attribute__((aligned(16))) bf16_t Ks[32 * 192];
  __shared__ __attribute__((aligned(16))) bf16_t Vs[128 * 32];   // [col][key]
  __shared__ float Sbuf[4][16][32];
  __shared__ __attribute__((aligned(16))) bf16_t Pbuf[4][16 * 32];
  __shared__ float alphaBuf[4][16];
  __shared__ float lBuf[4][16];

  const int tid  = threadIdx.x;
  const int wave = tid >> 5, lane = tid & 31;
  const int lm   = lane & 15, lhk = lane >> 4;
  const int qt = blockIdx.x, h = blockIdx.y, b = blockIdx.z;
  const int q0 = qt * 64;
  const size_t tb = (size_t)b * S;
  const float scale = rsqrtf(192.f);

  // load + scale Q tile (64 x 192)
#pragma unroll
  for (int it = 0; it < 24; ++it) {
    int f4 = tid + it * 128;               // 0..3071
    int r  = f4 / 48;
    int c  = (f4 % 48) * 4;
    float4 v = *(const float4*)(q + ((tb + q0 + r) * 16 + h) * 192 + c);
    bf16_t* d = &Qs[r * 192 + c];
    d[0] = f2bf(v.x * scale); d[1] = f2bf(v.y * scale);
    d[2] = f2bf(v.z * scale); d[3] = f2bf(v.w * scale);
  }

  v8f o[8];
#pragma unroll
  for (int jn = 0; jn < 8; ++jn)
#pragma unroll
    for (int q2 = 0; q2 < 8; ++q2) o[jn][q2] = 0.f;
  float mrow = -3.0e38f, lrow = 0.f;       // lanes 0..15 own row (wave*16+lm)

  const int kmax = q0 + 63;
  for (int k0 = 0; k0 <= kmax; k0 += 32) {
    // --- K tile: 32 x 192 (nope from kvp, pe from kpe) ---
#pragma unroll
    for (int it = 0; it < 12; ++it) {
      int f4 = tid + it * 128;             // 0..1535
      int r  = f4 / 48;
      int c  = (f4 % 48) * 4;
      float4 v;
      if (c < 128) v = *(const float4*)(kvp + ((tb + k0 + r) * 16 + h) * 256 + c);
      else         v = *(const float4*)(kpe + (tb + k0 + r) * 64 + (c - 128));
      bf16_t* d = &Ks[r * 192 + c];
      d[0] = f2bf(v.x); d[1] = f2bf(v.y); d[2] = f2bf(v.z); d[3] = f2bf(v.w);
    }
    // --- V tile transposed: Vs[col][key], 128 x 32 ---
#pragma unroll
    for (int it = 0; it < 8; ++it) {
      int f4 = tid + it * 128;             // 0..1023
      int r  = f4 >> 5;                    // key
      int c  = (f4 & 31) * 4;              // v col
      float4 v = *(const float4*)(kvp + ((tb + k0 + r) * 16 + h) * 256 + 128 + c);
      Vs[(c + 0) * 32 + r] = f2bf(v.x);
      Vs[(c + 1) * 32 + r] = f2bf(v.y);
      Vs[(c + 2) * 32 + r] = f2bf(v.z);
      Vs[(c + 3) * 32 + r] = f2bf(v.w);
    }
    __syncthreads();

    // --- scores S = Q · K^T (16x32 per wave) ---
    v8f sacc[2];
#pragma unroll
    for (int j = 0; j < 2; ++j)
#pragma unroll
      for (int q2 = 0; q2 < 8; ++q2) sacc[j][q2] = 0.f;
#pragma unroll
    for (int d = 0; d < 192; d += 32) {
      const bf16_t* ap = &Qs[(wave * 16 + lm) * 192 + d + lhk * 8];
      v16bf af = cat16(*(const v8bf*)ap, *(const v8bf*)(ap + 16));
#pragma unroll
      for (int j = 0; j < 2; ++j) {
        const bf16_t* bp = &Ks[(j * 16 + lm) * 192 + d + lhk * 8];
        v16bf bf_ = cat16(*(const v8bf*)bp, *(const v8bf*)(bp + 16));
        sacc[j] = WMMA_BF16(af, bf_, sacc[j]);
      }
    }
#pragma unroll
    for (int j = 0; j < 2; ++j)
#pragma unroll
      for (int q2 = 0; q2 < 8; ++q2)
        Sbuf[wave][lhk * 8 + q2][j * 16 + lm] = sacc[j][q2];
    __syncthreads();

    // --- online softmax update (lanes 0..15: one query row each) ---
    if (lane < 16) {
      const int qi = q0 + wave * 16 + lm;
      float mnew = mrow;
      for (int c = 0; c < 32; ++c) {
        float s = (k0 + c <= qi) ? Sbuf[wave][lm][c] : -3.0e38f;
        mnew = fmaxf(mnew, s);
      }
      float alpha = __expf(mrow - mnew);    // 0 if first real tile, 1 if unchanged
      float sum = 0.f;
      for (int c = 0; c < 32; ++c) {
        float s = (k0 + c <= qi) ? Sbuf[wave][lm][c] : -3.0e38f;
        float p = (s > -1.0e38f) ? __expf(s - mnew) : 0.f;
        Pbuf[wave][lm * 32 + c] = f2bf(p);
        sum += p;
      }
      mrow = mnew;
      lrow = lrow * alpha + sum;
      alphaBuf[wave][lm] = alpha;
    }
    __syncthreads();

    // --- rescale O, then O += P · V ---
#pragma unroll
    for (int jn = 0; jn < 8; ++jn)
#pragma unroll
      for (int q2 = 0; q2 < 8; ++q2)
        o[jn][q2] *= alphaBuf[wave][lhk * 8 + q2];

    const bf16_t* pp = &Pbuf[wave][lm * 32 + lhk * 8];
    v16bf pf = cat16(*(const v8bf*)pp, *(const v8bf*)(pp + 16));
#pragma unroll
    for (int jn = 0; jn < 8; ++jn) {
      const bf16_t* vp = &Vs[(jn * 16 + lm) * 32 + lhk * 8];
      v16bf vf = cat16(*(const v8bf*)vp, *(const v8bf*)(vp + 16));
      o[jn] = WMMA_BF16(pf, vf, o[jn]);
    }
    __syncthreads();
  }

  if (lane < 16) lBuf[wave][lm] = lrow;
  __syncthreads();
#pragma unroll
  for (int jn = 0; jn < 8; ++jn)
#pragma unroll
    for (int q2 = 0; q2 < 8; ++q2) {
      int r = wave * 16 + lhk * 8 + q2;
      float inv = 1.f / lBuf[wave][lhk * 8 + q2];
      out[((tb + q0 + r) * 16 + h) * 128 + jn * 16 + lm] = o[jn][q2] * inv;
    }
}

// ---------------------------------------------------------------------------
// Elementwise / small kernels
// ---------------------------------------------------------------------------
__global__ __launch_bounds__(256) void rmsnorm_k(
    const float* __restrict__ x, const float* __restrict__ w, float* __restrict__ out,
    int cols, int ldin, int ldout)
{
  __shared__ float red[256];
  __shared__ float snorm;
  const int row = blockIdx.x;
  const float* xr = x + (size_t)row * ldin;
  float s = 0.f;
  for (int c = threadIdx.x; c < cols; c += 256) { float v = xr[c]; s += v * v; }
  red[threadIdx.x] = s; __syncthreads();
  for (int st = 128; st > 0; st >>= 1) {
    if (threadIdx.x < st) red[threadIdx.x] += red[threadIdx.x + st];
    __syncthreads();
  }
  if (threadIdx.x == 0) snorm = rsqrtf(red[0] / (float)cols + 1e-6f);
  __syncthreads();
  float inv = snorm;
  float* orow = out + (size_t)row * ldout;
  for (int c = threadIdx.x; c < cols; c += 256) orow[c] = xr[c] * inv * w[c];
}

__global__ __launch_bounds__(128) void rope_k(
    float* __restrict__ q, const float* __restrict__ kv, float* __restrict__ kpe,
    const float* __restrict__ fc, int S)
{
  const int t = blockIdx.x;
  const int s = t % S;
  for (int p = threadIdx.x; p < 544; p += 128) {
    if (p < 512) {
      int hh = p >> 5, i = p & 31;
      float c = fc[(s * 32 + i) * 2 + 0], sn = fc[(s * 32 + i) * 2 + 1];
      float* ptr = q + ((size_t)t * 16 + hh) * 192 + 128 + 2 * i;
      float a = ptr[0], b = ptr[1];
      ptr[0] = a * c - b * sn;
      ptr[1] = a * sn + b * c;
    } else {
      int i = p - 512;
      float c = fc[(s * 32 + i) * 2 + 0], sn = fc[(s * 32 + i) * 2 + 1];
      const float* sp = kv + (size_t)t * 576 + 512 + 2 * i;
      float a = sp[0], b = sp[1];
      kpe[(size_t)t * 64 + 2 * i]     = a * c - b * sn;
      kpe[(size_t)t * 64 + 2 * i + 1] = a * sn + b * c;
    }
  }
}

__global__ void add2_k(const float* __restrict__ a, const float* __restrict__ b,
                       float* __restrict__ o, size_t n) {
  size_t i = (size_t)blockIdx.x * blockDim.x + threadIdx.x;
  if (i < n) o[i] = a[i] + b[i];
}
__global__ void zero_f32(float* p, size_t n) {
  size_t i = (size_t)blockIdx.x * blockDim.x + threadIdx.x;
  if (i < n) p[i] = 0.f;
}
__global__ void zero_i32(int* p, int n) {
  int i = blockIdx.x * blockDim.x + threadIdx.x;
  if (i < n) p[i] = 0;
}

__global__ __launch_bounds__(256) void moe_gate(
    const float* __restrict__ xn2, const float* __restrict__ gate_w,
    int* __restrict__ topi, float* __restrict__ topw, int H)
{
  __shared__ float red[256][8];
  const int t = blockIdx.x;
  float acc[8];
#pragma unroll
  for (int e = 0; e < 8; ++e) acc[e] = 0.f;
  for (int hh = threadIdx.x; hh < H; hh += 256) {
    float v = xn2[(size_t)t * H + hh];
    const float* g = gate_w + (size_t)hh * 8;
#pragma unroll
    for (int e = 0; e < 8; ++e) acc[e] += v * g[e];
  }
#pragma unroll
  for (int e = 0; e < 8; ++e) red[threadIdx.x][e] = acc[e];
  __syncthreads();
  for (int st = 128; st > 0; st >>= 1) {
    if (threadIdx.x < st)
#pragma unroll
      for (int e = 0; e < 8; ++e) red[threadIdx.x][e] += red[threadIdx.x + st][e];
    __syncthreads();
  }
  if (threadIdx.x == 0) {
    float lg[8]; float mx = -3.0e38f;
    for (int e = 0; e < 8; ++e) { lg[e] = red[0][e]; mx = fmaxf(mx, lg[e]); }
    float se = 0.f;
    for (int e = 0; e < 8; ++e) { lg[e] = __expf(lg[e] - mx); se += lg[e]; }
    for (int e = 0; e < 8; ++e) lg[e] /= se;
    int i0 = 0;
    for (int e = 1; e < 8; ++e) if (lg[e] > lg[i0]) i0 = e;
    int i1 = (i0 == 0) ? 1 : 0;
    for (int e = 0; e < 8; ++e) if (e != i0 && lg[e] > lg[i1]) i1 = e;
    float p0 = lg[i0], p1 = lg[i1], nrm = p0 + p1;
    topi[2 * t] = i0; topi[2 * t + 1] = i1;
    topw[2 * t] = p0 / nrm; topw[2 * t + 1] = p1 / nrm;
  }
}

__global__ void moe_count(const int* __restrict__ topi, int* __restrict__ counts, int n) {
  int i = blockIdx.x * blockDim.x + threadIdx.x;
  if (i < n) atomicAdd(&counts[topi[i]], 1);
}
__global__ void moe_scan(const int* __restrict__ counts, int* __restrict__ offsets,
                         int* __restrict__ cursor) {
  if (blockIdx.x == 0 && threadIdx.x == 0) {
    int s = 0;
    for (int e = 0; e < 8; ++e) { offsets[e] = s; s += counts[e]; cursor[e] = 0; }
  }
}
__global__ void moe_fill(const int* __restrict__ topi, const float* __restrict__ topw,
                         const int* __restrict__ offsets, int* __restrict__ cursor,
                         int* __restrict__ atok, float* __restrict__ aw, int T) {
  int t = blockIdx.x * blockDim.x + threadIdx.x;
  if (t < T) {
    for (int k = 0; k < 2; ++k) {
      int e = topi[2 * t + k];
      int pos = atomicAdd(&cursor[e], 1);
      int s = offsets[e] + pos;
      atok[s] = t;
      aw[s] = topw[2 * t + k];
    }
  }
}

// ---------------------------------------------------------------------------
// Launcher
// ---------------------------------------------------------------------------
extern "C" void kernel_launch(void* const* d_in, const int* in_sizes, int n_in,
                              void* d_out, int out_size, void* d_ws, size_t ws_size,
                              hipStream_t stream)
{
  const float* x      = (const float*)d_in[0];
  const float* fc     = (const float*)d_in[1];
  const float* norm1w = (const float*)d_in[2];
  const float* norm2w = (const float*)d_in[3];
  const float* wq_a   = (const float*)d_in[4];
  const float* qnw    = (const float*)d_in[5];
  const float* wq_b   = (const float*)d_in[6];
  const float* wkv_a  = (const float*)d_in[7];
  const float* kvnw   = (const float*)d_in[8];
  const float* wkv_b  = (const float*)d_in[9];
  const float* wo     = (const float*)d_in[10];
  const float* gatew  = (const float*)d_in[11];
  const float* w1     = (const float*)d_in[12];
  const float* w3     = (const float*)d_in[13];
  const float* w2     = (const float*)d_in[14];
  const float* w1s    = (const float*)d_in[15];
  const float* w3s    = (const float*)d_in[16];
  const float* w2s    = (const float*)d_in[17];

  const int S = 2048, H = 2048, T = 2 * S;     // B=2
  const int QR = 1536, MI = 1408, SI = 5632;

  char* wp = (char*)d_ws;
  auto alloc = [&](size_t bytes) -> void* {
    void* p = (void*)wp;
    wp += (bytes + 255) & ~(size_t)255;
    return p;
  };

  float*  xn1   = (float*)alloc((size_t)T * H * 4);
  float*  qlat  = (float*)alloc((size_t)T * QR * 4);
  float*  qlatn = (float*)alloc((size_t)T * QR * 4);
  float*  qbuf  = (float*)alloc((size_t)T * 3072 * 4);
  float*  kvbuf = (float*)alloc((size_t)T * 576 * 4);
  float*  kvn   = (float*)alloc((size_t)T * 512 * 4);
  float*  kvp   = (float*)alloc((size_t)T * 4096 * 4);
  float*  kpe   = (float*)alloc((size_t)T * 64 * 4);
  float*  aout  = (float*)alloc((size_t)T * 2048 * 4);
  float*  aproj = (float*)alloc((size_t)T * 2048 * 4);
  float*  x1    = (float*)alloc((size_t)T * H * 4);
  float*  xn2   = (float*)alloc((size_t)T * H * 4);
  int*    topi  = (int*)alloc((size_t)2 * T * 4);
  float*  topw  = (float*)alloc((size_t)2 * T * 4);
  int*    counts  = (int*)alloc(8 * 4);
  int*    offsets = (int*)alloc(8 * 4);
  int*    cursor  = (int*)alloc(8 * 4);
  int*    atok  = (int*)alloc((size_t)2 * T * 4);
  float*  aw    = (float*)alloc((size_t)2 * T * 4);
  bf16_t* h1    = (bf16_t*)alloc((size_t)2 * T * MI * 2);
  bf16_t* h1s   = (bf16_t*)alloc((size_t)T * SI * 2);
  float*  accb  = (float*)alloc((size_t)T * H * 4);

  // ---- attention branch ----
  rmsnorm_k<<<T, 256, 0, stream>>>(x, norm1w, xn1, H, H, H);
  gemm_nn<<<dim3(12, 32), 256, 0, stream>>>(xn1, wq_a, qlat, T, QR, H, H, QR, QR);
  rmsnorm_k<<<T, 256, 0, stream>>>(qlat, qnw, qlatn, QR, QR, QR);
  gemm_nn<<<dim3(24, 32), 256, 0, stream>>>(qlatn, wq_b, qbuf, T, 3072, QR, QR, 3072, 3072);
  gemm_nn<<<dim3(5, 32), 256, 0, stream>>>(xn1, wkv_a, kvbuf, T, 576, H, H, 576, 576);
  rmsnorm_k<<<T, 256, 0, stream>>>(kvbuf, kvnw, kvn, 512, 576, 512);
  gemm_nn<<<dim3(32, 32), 256, 0, stream>>>(kvn, wkv_b, kvp, T, 4096, 512, 512, 4096, 4096);
  rope_k<<<T, 128, 0, stream>>>(qbuf, kvbuf, kpe, fc, S);
  mla_attention<<<dim3(32, 16, 2), 128, 0, stream>>>(qbuf, kvp, kpe, aout, S);
  gemm_nn<<<dim3(16, 32), 256, 0, stream>>>(aout, wo, aproj, T, H, 2048, 2048, H, H);
  add2_k<<<(int)(((size_t)T * H + 255) / 256), 256, 0, stream>>>(x, aproj, x1, (size_t)T * H);

  // ---- MoE branch ----
  rmsnorm_k<<<T, 256, 0, stream>>>(x1, norm2w, xn2, H, H, H);
  moe_gate<<<T, 256, 0, stream>>>(xn2, gatew, topi, topw, H);
  zero_i32<<<1, 32, 0, stream>>>(counts, 8);
  moe_count<<<(2 * T + 255) / 256, 256, 0, stream>>>(topi, counts, 2 * T);
  moe_scan<<<1, 1, 0, stream>>>(counts, offsets, cursor);
  moe_fill<<<(T + 255) / 256, 256, 0, stream>>>(topi, topw, offsets, cursor, atok, aw, T);
  zero_f32<<<(int)(((size_t)T * H + 255) / 256), 256, 0, stream>>>(accb, (size_t)T * H);

  // routed experts: gathered dual GEMM -> scatter-accumulate down-proj
  moe_gemm1<<<dim3(11, 64, 8), 256, 0, stream>>>(xn2, w1, w3, h1, atok, counts, offsets,
                                                 T, H, MI, (size_t)H * MI);
  moe_gemm2<<<dim3(16, 32, 8), 256, 0, stream>>>(h1, w2, accb, atok, aw, counts, offsets,
                                                 T, MI, H, (size_t)MI * H);
  // shared expert: same kernels, identity gather, weight 1
  moe_gemm1<<<dim3(44, 64, 1), 256, 0, stream>>>(xn2, w1s, w3s, h1s, nullptr, nullptr, nullptr,
                                                 T, H, SI, 0);
  moe_gemm2<<<dim3(16, 32, 1), 256, 0, stream>>>(h1s, w2s, accb, nullptr, nullptr, nullptr, nullptr,
                                                 T, SI, H, 0);

  add2_k<<<(int)(((size_t)T * H + 255) / 256), 256, 0, stream>>>(x1, accb, (float*)d_out,
                                                                 (size_t)T * H);
}
